// Head_77008763617429
// MI455X (gfx1250) — matrix operand order repaired and checked by reading, not compiled, not run on hardware
//
#include <hip/hip_runtime.h>
#include <cstddef>
#include <cstdint>

// ---------------- problem constants (from reference) ----------------
#define EDIM 1024
#define HDIM 64
#define BATCH 8
#define TLEN 2048

typedef __attribute__((ext_vector_type(16))) __bf16 v16bf;
typedef __attribute__((ext_vector_type(8)))  float  v8f;
typedef __attribute__((ext_vector_type(4)))  unsigned int v4u;
typedef __attribute__((ext_vector_type(4)))  int    v4i;
typedef __attribute__((ext_vector_type(8)))  int    v8i;

// ---------------- WMMA fragment loaders (per CDNA5 ISA layouts) ----------------
// A-matrix 16x32 bf16: lane(half,m): K = kbase + half*8 .. +7  and  +16 .. +23
__device__ __forceinline__ v16bf load_a_frag(const __bf16* base, int stride, int lane) {
  const int half = lane >> 4;
  const int m    = lane & 15;
  const __bf16* p = base + m * stride + half * 8;
  union { v16bf v; uint4 u[2]; } t;
  t.u[0] = *(const uint4*)(p);
  t.u[1] = *(const uint4*)(p + 16);
  return t.v;
}

// B-matrix 32x16 bf16 from column-major staging: lane(half,n): K = kbase + half*8 .. +7  and  +16 .. +23
__device__ __forceinline__ v16bf load_b_frag(const __bf16* baseT, int stride, int lane) {
  const int half = lane >> 4;
  const int n    = lane & 15;
  const __bf16* p = baseT + n * stride + half * 8;
  union { v16bf v; uint4 u[2]; } t;
  t.u[0] = *(const uint4*)(p);
  t.u[1] = *(const uint4*)(p + 16);
  return t.v;
}

__device__ __forceinline__ v8f wmma_bf16(v16bf a, v16bf b, v8f c) {
  return __builtin_amdgcn_wmma_f32_16x16x32_bf16(false, a, false, b, (short)0, c, false, false);
}

// ---------------- Tensor Data Mover: stage a 32x64 bf16 tile to LDS ----------------
// D# per ISA 8.3/8.4: group0 {count=1, lds_addr, global_addr[56:0], type=2},
// group1 {data_size=2B, tensor_dim0=64, tensor_dim1=2048, tile 64x32, stride0=64}.
__device__ __forceinline__ void tdm_load_tile32x64(const __bf16* gsrc, __bf16* ldst) {
  const unsigned lds_addr = (unsigned)(uintptr_t)ldst;            // generic low 32 = LDS byte addr
  const unsigned long long ga = (unsigned long long)(uintptr_t)gsrc;
  v4u g0;
  g0.x = 1u;                                                      // count=1, user descriptor
  g0.y = lds_addr;                                                // lds_addr [63:32]
  g0.z = (unsigned)ga;                                            // global_addr [95:64]
  g0.w = (unsigned)((ga >> 32) & 0x01FFFFFFu) | (2u << 30);       // addr[56:32] | type=2
  v8i g1;
  g1[0] = (int)(1u << 16);          // data_size=1 (2 bytes)
  g1[1] = (int)(64u << 16);         // tensor_dim0 = 64  (bits 79:48)
  g1[2] = (int)(2048u << 16);       // tensor_dim1 = 2048 (bits 111:80)
  g1[3] = (int)(64u << 16);         // tile_dim0 = 64 (bits 127:112)
  g1[4] = 32;                       // tile_dim1 = 32 (bits 143:128)
  g1[5] = 64;                       // tensor_dim0_stride = 64 (bits 191:160)
  g1[6] = 0;
  g1[7] = 0;
  const v4i z4 = (v4i){0, 0, 0, 0};
#if defined(__clang_major__) && (__clang_major__ >= 23)
  const v8i z8 = (v8i){0, 0, 0, 0, 0, 0, 0, 0};
  __builtin_amdgcn_tensor_load_to_lds(g0, g1, z4, z4, z8, 0);
#else
  __builtin_amdgcn_tensor_load_to_lds(g0, g1, z4, z4, 0);
#endif
}

// ======================================================================
// Kernel 1: fused QKV projection.  [B*T,1024] x [1024,64] (x3) -> bf16
// 256 threads = 8 waves; block tile = 128 rows; K streamed in chunks of 64.
// ======================================================================
#define KC 64

__global__ __launch_bounds__(256) void qkv_proj_kernel(
    const float* __restrict__ x,
    const float* __restrict__ Wq, const float* __restrict__ bq,
    const float* __restrict__ Wk, const float* __restrict__ bk,
    const float* __restrict__ Wv, const float* __restrict__ bv,
    __bf16* __restrict__ qh, __bf16* __restrict__ kh, __bf16* __restrict__ vh)
{
  __shared__ __align__(16) __bf16 lds_x[128 * KC];        // 16 KB, [row][k]
  __shared__ __align__(16) __bf16 lds_w[3 * HDIM * KC];   // 24 KB, [m][n][k] (transposed)

  const int tid  = threadIdx.x;
  const int wave = tid >> 5;
  const int lane = tid & 31;
  const int half = lane >> 4;
  const int ln   = lane & 15;
  const int row0 = blockIdx.x * 128;

  v8f acc[3][4];
#pragma unroll
  for (int m = 0; m < 3; ++m)
#pragma unroll
    for (int j = 0; j < 4; ++j)
      acc[m][j] = (v8f){0.f,0.f,0.f,0.f,0.f,0.f,0.f,0.f};

  for (int k0 = 0; k0 < EDIM; k0 += KC) {
    // stage x chunk: float4 coalesced reads, packed 8B LDS writes
    for (int i = tid; i < (128 * KC) / 4; i += 256) {
      const int r  = i >> 4;              // 16 float4 per row
      const int c4 = (i & 15) * 4;
      const float4 f = *(const float4*)&x[(size_t)(row0 + r) * EDIM + k0 + c4];
      union { __bf16 h[4]; uint2 u; } t;
      t.h[0] = (__bf16)f.x; t.h[1] = (__bf16)f.y;
      t.h[2] = (__bf16)f.z; t.h[3] = (__bf16)f.w;
      *(uint2*)&lds_x[r * KC + c4] = t.u;
    }
    // stage W chunks transposed: float4 coalesced along n, scatter [n][k]
    for (int i = tid; i < (3 * HDIM * KC) / 4; i += 256) {
      const int m   = i / ((HDIM * KC) / 4);
      const int rem = i - m * ((HDIM * KC) / 4);
      const int c   = rem >> 4;           // k within chunk
      const int n4  = (rem & 15) * 4;     // group of 4 output columns
      const float* Wm = (m == 0) ? Wq : ((m == 1) ? Wk : Wv);
      const float4 f = *(const float4*)&Wm[(size_t)(k0 + c) * HDIM + n4];
      __bf16* wp = &lds_w[m * (HDIM * KC) + n4 * KC + c];
      wp[0 * KC] = (__bf16)f.x; wp[1 * KC] = (__bf16)f.y;
      wp[2 * KC] = (__bf16)f.z; wp[3 * KC] = (__bf16)f.w;
    }
    __syncthreads();

#pragma unroll
    for (int kk = 0; kk < KC; kk += 32) {
      const v16bf a = load_a_frag(&lds_x[(wave * 16) * KC + kk], KC, lane);
#pragma unroll
      for (int m = 0; m < 3; ++m) {
        v16bf bfr[4];                     // batch loads -> one wait, 4-WMMA clause
#pragma unroll
        for (int j = 0; j < 4; ++j)
          bfr[j] = load_b_frag(&lds_w[m * (HDIM * KC) + (j * 16) * KC + kk], KC, lane);
#pragma unroll
        for (int j = 0; j < 4; ++j)
          acc[m][j] = wmma_bf16(a, bfr[j], acc[m][j]);
      }
    }
    __syncthreads();
  }

  // bias add (f32) + bf16 store.  C-layout: row = r + 8*half, col = j*16 + ln
#pragma unroll
  for (int m = 0; m < 3; ++m) {
    const float* bm = (m == 0) ? bq : ((m == 1) ? bk : bv);
    __bf16* om = (m == 0) ? qh : ((m == 1) ? kh : vh);
#pragma unroll
    for (int j = 0; j < 4; ++j) {
      const int col = j * 16 + ln;
      const float bias = bm[col];
#pragma unroll
      for (int r = 0; r < 8; ++r) {
        const int row = row0 + wave * 16 + r + 8 * half;
        om[(size_t)row * HDIM + col] = (__bf16)(acc[m][j][r] + bias);
      }
    }
  }
}

// ======================================================================
// Kernel 2: causal flash attention, TDM double-buffered k/v staging.
// 128 threads = 4 waves; block = 64 query rows (16 per wave); key step = 32.
// ======================================================================
__global__ __launch_bounds__(128) void flash_attn_kernel(
    const __bf16* __restrict__ qh, const __bf16* __restrict__ kh,
    const __bf16* __restrict__ vh, float* __restrict__ out)
{
  __shared__ __align__(16) __bf16 lds_k[2][32 * HDIM];    // [buf][s][h]   8 KB
  __shared__ __align__(16) __bf16 lds_vraw[2][32 * HDIM]; // [buf][s][h]   8 KB
  __shared__ __align__(16) __bf16 lds_vt[HDIM * 32];      // [h][s]        4 KB
  __shared__ __align__(16) __bf16 lds_p[4][16 * 32];      // per-wave P    4 KB

  const int b    = blockIdx.y;
  const int q0   = blockIdx.x * 64;
  const int wave = threadIdx.x >> 5;
  const int lane = threadIdx.x & 31;
  const int half = lane >> 4;
  const int ln   = lane & 15;
  const int qrow = q0 + wave * 16;

  // resident q A-fragments (h-chunks 0 and 1)
  const __bf16* qptr = qh + ((size_t)b * TLEN + qrow) * HDIM;
  const v16bf qa0 = load_a_frag(qptr + 0,  HDIM, lane);
  const v16bf qa1 = load_a_frag(qptr + 32, HDIM, lane);

  v8f o[4];
#pragma unroll
  for (int j = 0; j < 4; ++j) o[j] = (v8f){0.f,0.f,0.f,0.f,0.f,0.f,0.f,0.f};
  float mrow[8], lrow[8];
#pragma unroll
  for (int r = 0; r < 8; ++r) { mrow[r] = -1e30f; lrow[r] = 0.f; }

  const float scale = 0.125f;  // 1/sqrt(64)
  const int nT = (q0 + 64) >> 5;              // causal tile count for this block

  // prologue: TDM prefetch of tile 0 (wave 0 drives the DMA engine)
  if (wave == 0) {
    const size_t off0 = ((size_t)b * TLEN) * HDIM;
    tdm_load_tile32x64(kh + off0, &lds_k[0][0]);
    tdm_load_tile32x64(vh + off0, &lds_vraw[0][0]);
  }

  for (int t = 0; t < nT; ++t) {
    const int cur = t & 1;
    const int s0  = t * 32;

    if (wave == 0) {
      if (t + 1 < nT) {
        // stream tile t+1 while we compute tile t
        const size_t offn = ((size_t)b * TLEN + (t + 1) * 32) * HDIM;
        tdm_load_tile32x64(kh + offn, &lds_k[cur ^ 1][0]);
        tdm_load_tile32x64(vh + offn, &lds_vraw[cur ^ 1][0]);
        __builtin_amdgcn_s_wait_tensorcnt(2);   // retire tile t (in-order TDM)
      } else {
        __builtin_amdgcn_s_wait_tensorcnt(0);
      }
    }
    __syncthreads();                            // tile t visible to all waves

    // transpose v tile in LDS: [s][h] -> [h][s] (reads contiguous)
    for (int i = threadIdx.x; i < 32 * HDIM; i += 128) {
      const int s = i >> 6, h = i & 63;
      lds_vt[h * 32 + s] = lds_vraw[cur][s * HDIM + h];
    }
    __syncthreads();

    if (s0 <= qrow + 15) {                      // wave-uniform causal skip
      // ---- S = q . k^T  (16 x 32 scores): batch 4 B-frags then 4 WMMAs ----
      v8f sacc[2];
      sacc[0] = (v8f){0.f,0.f,0.f,0.f,0.f,0.f,0.f,0.f};
      sacc[1] = (v8f){0.f,0.f,0.f,0.f,0.f,0.f,0.f,0.f};
      v16bf kb[2][2];
#pragma unroll
      for (int hc = 0; hc < 2; ++hc)
#pragma unroll
        for (int j = 0; j < 2; ++j)
          kb[hc][j] = load_b_frag(&lds_k[cur][(j * 16) * HDIM + hc * 32], HDIM, lane);
#pragma unroll
      for (int hc = 0; hc < 2; ++hc) {
        const v16bf a = (hc == 0) ? qa0 : qa1;
#pragma unroll
        for (int j = 0; j < 2; ++j)
          sacc[j] = wmma_bf16(a, kb[hc][j], sacc[j]);
      }

      // ---- online softmax (per query row; 16-lane butterflies) ----
#pragma unroll
      for (int r = 0; r < 8; ++r) {
        const int qi = qrow + r + 8 * half;
        float v0 = sacc[0][r] * scale;
        float v1 = sacc[1][r] * scale;
        if (s0 + ln      > qi) v0 = -1e30f;
        if (s0 + 16 + ln > qi) v1 = -1e30f;
        float mx = fmaxf(v0, v1);
#pragma unroll
        for (int off = 1; off < 16; off <<= 1)
          mx = fmaxf(mx, __shfl_xor(mx, off, 32));
        const float mnew  = fmaxf(mrow[r], mx);
        const float alpha = __expf(mrow[r] - mnew);
        const float p0 = __expf(v0 - mnew);
        const float p1 = __expf(v1 - mnew);
        float sum = p0 + p1;
#pragma unroll
        for (int off = 1; off < 16; off <<= 1)
          sum += __shfl_xor(sum, off, 32);
        lrow[r] = lrow[r] * alpha + sum;
        mrow[r] = mnew;
#pragma unroll
        for (int j = 0; j < 4; ++j) o[j][r] *= alpha;
        const int prow = r + 8 * half;
        lds_p[wave][prow * 32 + ln]      = (__bf16)p0;
        lds_p[wave][prow * 32 + 16 + ln] = (__bf16)p1;
      }

      // CDNA5 split counters: drain DS stores before re-reading P as A-frag
      asm volatile("s_wait_dscnt 0" ::: "memory");

      // ---- O += P @ V : batch 4 B-frags then 4-WMMA chain ----
      const v16bf pa = load_a_frag(&lds_p[wave][0], 32, lane);
      v16bf vtf[4];
#pragma unroll
      for (int j = 0; j < 4; ++j)
        vtf[j] = load_b_frag(&lds_vt[(j * 16) * 32], 32, lane);
#pragma unroll
      for (int j = 0; j < 4; ++j)
        o[j] = wmma_bf16(pa, vtf[j], o[j]);
    }
    __syncthreads();                            // buffer `cur` free for TDM reuse
  }

  // ---- normalize and store f32 output ----
  float* op = out + ((size_t)b * TLEN + qrow) * HDIM;
#pragma unroll
  for (int j = 0; j < 4; ++j)
#pragma unroll
    for (int r = 0; r < 8; ++r) {
      const int row = r + 8 * half;
      op[(size_t)row * HDIM + j * 16 + ln] = o[j][r] / lrow[r];
    }
}

// ======================================================================
extern "C" void kernel_launch(void* const* d_in, const int* in_sizes, int n_in,
                              void* d_out, int out_size, void* d_ws, size_t ws_size,
                              hipStream_t stream) {
  const float* x  = (const float*)d_in[0];
  const float* Wq = (const float*)d_in[1];
  const float* bq = (const float*)d_in[2];
  const float* Wk = (const float*)d_in[3];
  const float* bk = (const float*)d_in[4];
  const float* Wv = (const float*)d_in[5];
  const float* bv = (const float*)d_in[6];
  float* out = (float*)d_out;

  // workspace: q/k/v in bf16, 3 * B*T*H * 2B = 6 MB
  const size_t qkv_elems = (size_t)BATCH * TLEN * HDIM;
  __bf16* qh = (__bf16*)d_ws;
  __bf16* kh = qh + qkv_elems;
  __bf16* vh = kh + qkv_elems;

  const int rows = BATCH * TLEN;                       // 16384
  qkv_proj_kernel<<<dim3(rows / 128), dim3(256), 0, stream>>>(
      x, Wq, bq, Wk, bk, Wv, bv, qh, kh, vh);

  flash_attn_kernel<<<dim3(TLEN / 64, BATCH), dim3(128), 0, stream>>>(
      qh, kh, vh, out);
}
